// CutCrossEntropyLoss_66005057405073
// MI455X (gfx1250) — compile-verified, matrix operand order checked by reading, not run on hardware
//
#include <hip/hip_runtime.h>
#include <hip/hip_bf16.h>

#define IGNORE_INDEX (-100)
#define D_DIM     2048
#define SROWS     2047            // valid rows of e = embeddings[:, :-1, :]
#define SROWS_PAD 2048            // padded with a zero row -> no guards in GEMM
#define V_DIM     128000
#define VT        64              // vocab columns per block (W stripe in LDS)
#define NVB       (V_DIM / VT)    // 2000 blocks
#define KTILE     32
#define NKI       (D_DIM / KTILE) // 64
#define PITCH_H   (D_DIM + 8)     // LDS pitch in bf16 elems: stride 4 banks/lane
#define PITCH_B   (PITCH_H * 2)   // 4112 bytes, 16B aligned
#define BLOCK_T   512             // 16 waves: 4 col-subtiles x 4 row-groups
#define ROWS_PER_OUTER 256        // 4 row-groups x 4 row-tiles x 16
#define NOUT      (SROWS_PAD / ROWS_PER_OUTER)  // 8

typedef __attribute__((ext_vector_type(16))) __bf16 v16bf;
typedef __attribute__((ext_vector_type(8)))  float  v8f;

union Frag16 {
    unsigned u[8];
    uint4    q[2];
    v16bf    v;
};

// one v_perm_b32: {hi[31:16], lo[31:16]} (truncating fp32->bf16 pack)
__device__ __forceinline__ unsigned pk_bf16(float lo, float hi) {
    return __builtin_amdgcn_perm(__float_as_uint(hi), __float_as_uint(lo),
                                 0x07060302u);
}

// ---------------------------------------------------------------------------
// Kernel 0: embeddings fp32 -> bf16 (row 2047 zero-padded).
// ---------------------------------------------------------------------------
__global__ __launch_bounds__(256)
void cce_convert_emb(const float* __restrict__ emb,
                     unsigned short* __restrict__ Abf)  // [2048][2048] bf16
{
    const size_t idx = ((size_t)blockIdx.x * 256 + threadIdx.x) * 8;
    const int row = (int)(idx >> 11);
    uint4 v;
    if (row < SROWS) {
        const float4* g = (const float4*)(emb + idx);
        float4 p0 = g[0], p1 = g[1];
        v.x = pk_bf16(p0.x, p0.y);
        v.y = pk_bf16(p0.z, p0.w);
        v.z = pk_bf16(p1.x, p1.y);
        v.w = pk_bf16(p1.z, p1.w);
    } else {
        v = make_uint4(0u, 0u, 0u, 0u);
    }
    *(uint4*)(Abf + idx) = v;
}

// ---------------------------------------------------------------------------
// Kernel 1: each block owns a 64-col vocab stripe. Stage W stripe -> LDS
// (bf16, pitched, conflict-free b128 reads), then sweep all 2048 rows with
// bf16 WMMA; 16 waves (4 waves/SIMD) for latency hiding. Emits per-(row,
// block) online-softmax partials and the label logit.
// ---------------------------------------------------------------------------
__global__ __launch_bounds__(BLOCK_T)
void cce_gemm_partials(const float*          __restrict__ W,
                       const float*          __restrict__ bias,
                       const int*            __restrict__ labels,
                       const unsigned short* __restrict__ Abf,
                       float2*               __restrict__ partials,  // [2048][NVB]
                       float*                __restrict__ lab_logit) // [SROWS]
{
    __shared__ __align__(16) unsigned short wtile[VT * PITCH_H]; // 263168 B
    __shared__ float2 redbuf[16][64];                            // 8192 B

    const int tid  = threadIdx.x;
    const int lane = tid & 31;
    const int wave = tid >> 5;          // 0..15
    const int n    = lane & 15;
    const int half = lane >> 4;
    const int kg0  = half ? 8 : 0;      // ISA 16-bit fragment K pattern
    const int cw   = wave & 3;          // col subtile within stripe
    const int rg   = wave >> 2;         // row group (0..3)
    const int blockCol = blockIdx.x * VT;
    const int colN     = blockCol + cw * 16 + n;

    // ---- stage W stripe (64 cols x 2048 K) into LDS as bf16, once ----
    #pragma unroll 4
    for (int i = 0; i < (VT * D_DIM / 8) / BLOCK_T; ++i) {   // 32 iters
        const int ch = tid + BLOCK_T * i;
        const int c  = ch >> 8;          // 0..63 (256 chunks of 8 per col)
        const int kc = ch & 255;
        const float4* g = (const float4*)(W + (size_t)(blockCol + c) * D_DIM + kc * 8);
        float4 p0 = g[0], p1 = g[1];
        uint4 v;
        v.x = pk_bf16(p0.x, p0.y);
        v.y = pk_bf16(p0.z, p0.w);
        v.z = pk_bf16(p1.x, p1.y);
        v.w = pk_bf16(p1.z, p1.w);
        *(uint4*)((char*)wtile + (size_t)c * PITCH_B + kc * 16) = v;
    }
    __syncthreads();

    const char* lbBase = (const char*)wtile + (size_t)(cw * 16 + n) * PITCH_B + kg0 * 2;
    const float bia    = bias[colN];

    for (int oi = 0; oi < NOUT; ++oi) {
        const int rows0 = oi * ROWS_PER_OUTER + rg * 64;
        const unsigned short* aBase = Abf + (size_t)(rows0 + n) * D_DIM + kg0;

        v8f acc[4] = {v8f{}, v8f{}, v8f{}, v8f{}};

        #pragma unroll 4
        for (int ki = 0; ki < NKI; ++ki) {
            Frag16 b;
            b.q[0] = *(const uint4*)(lbBase + ki * 64);
            b.q[1] = *(const uint4*)(lbBase + ki * 64 + 32);
            #pragma unroll
            for (int rt = 0; rt < 4; ++rt) {
                Frag16 a;
                const unsigned short* ap = aBase + rt * 16 * D_DIM + ki * KTILE;
                a.q[0] = *(const uint4*)ap;
                a.q[1] = *(const uint4*)(ap + 16);
                acc[rt] = __builtin_amdgcn_wmma_f32_16x16x32_bf16(
                              false, a.v, false, b.v, (short)0, acc[rt],
                              false, false);
            }
        }

        // ---- epilogue: bias, label capture, 16-lane log-sum-exp ----
        #pragma unroll
        for (int rt = 0; rt < 4; ++rt) {
            #pragma unroll
            for (int r = 0; r < 8; ++r) {
                float x = acc[rt][r] + bia;
                const int lrow = rt * 16 + r + half * 8;
                const int grow = rows0 + lrow;
                if (grow < SROWS) {
                    const int lbl = labels[grow + 1];   // y = labels[:,1:]
                    if (lbl == colN) lab_logit[grow] = x;
                }
                float mm = x;
                #pragma unroll
                for (int mk = 1; mk <= 8; mk <<= 1)
                    mm = fmaxf(mm, __shfl_xor(mm, mk, 32));
                float e = __expf(x - mm);
                #pragma unroll
                for (int mk = 1; mk <= 8; mk <<= 1)
                    e += __shfl_xor(e, mk, 32);
                if (n == 0) redbuf[wave][lrow] = make_float2(mm, e);
            }
        }
        __syncthreads();

        // merge the 4 col-waves of each row group -> one partial per row
        if (tid < 256) {
            const int rloc = tid & 63;
            const int trg  = tid >> 6;      // row group 0..3
            float M = -INFINITY, S = 0.0f;
            #pragma unroll
            for (int c4 = 0; c4 < 4; ++c4) {
                float2 v = redbuf[trg * 4 + c4][rloc];
                float nm = fmaxf(M, v.x);
                S = S * __expf(M - nm) + v.y * __expf(v.x - nm);
                M = nm;
            }
            const int grow = oi * ROWS_PER_OUTER + trg * 64 + rloc;
            partials[(size_t)grow * NVB + blockIdx.x] = make_float2(M, S);
        }
        __syncthreads();
    }
}

// ---------------------------------------------------------------------------
// Kernel 2: one block per row, merge NVB partials -> nll[row].
// ---------------------------------------------------------------------------
__global__ __launch_bounds__(256)
void cce_row_reduce(const float2* __restrict__ partials,
                    const float*  __restrict__ lab_logit,
                    const int*    __restrict__ labels,
                    float*        __restrict__ nll)    // [SROWS]
{
    __shared__ float sM[256], sS[256];
    const int row = blockIdx.x;
    const float2* p = partials + (size_t)row * NVB;

    float M = -INFINITY, S = 0.0f;
    for (int b = threadIdx.x; b < NVB; b += 256) {
        float2 v = p[b];
        float nm = fmaxf(M, v.x);
        S = S * __expf(M - nm) + v.y * __expf(v.x - nm);
        M = nm;
    }
    sM[threadIdx.x] = M;
    sS[threadIdx.x] = S;
    __syncthreads();
    for (int off = 128; off > 0; off >>= 1) {
        if ((int)threadIdx.x < off) {
            float m2 = sM[threadIdx.x + off], s2 = sS[threadIdx.x + off];
            float m1 = sM[threadIdx.x],       s1 = sS[threadIdx.x];
            float nm = fmaxf(m1, m2);
            sS[threadIdx.x] = s1 * __expf(m1 - nm) + s2 * __expf(m2 - nm);
            sM[threadIdx.x] = nm;
        }
        __syncthreads();
    }
    if (threadIdx.x == 0) {
        const int lbl = labels[row + 1];
        nll[row] = (lbl != IGNORE_INDEX)
                     ? (sM[0] + logf(sS[0]) - lab_logit[row]) : 0.0f;
    }
}

// ---------------------------------------------------------------------------
// Kernel 3: masked mean over rows.
// ---------------------------------------------------------------------------
__global__ __launch_bounds__(256)
void cce_final(const float* __restrict__ nll,
               const int*   __restrict__ labels,
               float*       __restrict__ out)
{
    __shared__ float ssum[256];
    __shared__ int   scnt[256];
    float acc = 0.0f; int cnt = 0;
    for (int row = threadIdx.x; row < SROWS; row += 256) {
        acc += nll[row];
        cnt += (labels[row + 1] != IGNORE_INDEX) ? 1 : 0;
    }
    ssum[threadIdx.x] = acc;
    scnt[threadIdx.x] = cnt;
    __syncthreads();
    for (int off = 128; off > 0; off >>= 1) {
        if ((int)threadIdx.x < off) {
            ssum[threadIdx.x] += ssum[threadIdx.x + off];
            scnt[threadIdx.x] += scnt[threadIdx.x + off];
        }
        __syncthreads();
    }
    if (threadIdx.x == 0) {
        int denom = scnt[0] > 1 ? scnt[0] : 1;
        out[0] = ssum[0] / (float)denom;
    }
}

// ---------------------------------------------------------------------------
extern "C" void kernel_launch(void* const* d_in, const int* in_sizes, int n_in,
                              void* d_out, int out_size, void* d_ws, size_t ws_size,
                              hipStream_t stream) {
    (void)in_sizes; (void)n_in; (void)out_size; (void)ws_size;
    // inputs: logits, embeddings, classifier_weight, classifier_bias, labels, input_ids
    const float* emb    = (const float*)d_in[1];
    const float* W      = (const float*)d_in[2];
    const float* bias   = (const float*)d_in[3];
    const int*   labels = (const int*)d_in[4];

    char* ws = (char*)d_ws;
    const size_t PART_BYTES = (size_t)SROWS_PAD * NVB * sizeof(float2); // 32.77 MB
    float2*         partials  = (float2*)ws;
    float*          lab_logit = (float*)(ws + PART_BYTES);              // 8 KB slot
    float*          nll       = (float*)(ws + PART_BYTES + 8192);
    unsigned short* Abf       = (unsigned short*)(ws + PART_BYTES + 16384);

    cce_convert_emb<<<(SROWS_PAD * D_DIM) / (256 * 8), 256, 0, stream>>>(emb, Abf);
    cce_gemm_partials<<<NVB, BLOCK_T, 0, stream>>>(W, bias, labels, Abf,
                                                   partials, lab_logit);
    cce_row_reduce<<<SROWS, 256, 0, stream>>>(partials, lab_logit, labels, nll);
    cce_final<<<1, 256, 0, stream>>>(nll, labels, (float*)d_out);
}
